// Conv2d_738734375712
// MI455X (gfx1250) — matrix operand order repaired and checked by reading, not compiled, not run on hardware
//
#include <hip/hip_runtime.h>
#include <hip/hip_bf16.h>

// Conv2d 3x3 VALID stride-1 as implicit GEMM on BF16 WMMA, f32 accumulate.
// GEMM: D[o][pix],  M=256 (K_out), N=387200 (=32*110*110), K=1152 (=128*9).
// K ordering: k' = rs*128 + c (rs-major): each 32-wide K-step has fixed (r,s)
// and 32 consecutive input channels -> immediate-offset gather loads.
// Filter pre-converted once to bf16 wsA[o][k'] (590 KB in d_ws); the A tile is
// streamed into LDS with GLOBAL_LOAD_ASYNC_TO_LDS_B128 (no VGPR staging).

#define CIN    128
#define HIN    112
#define WIN    112
#define NBATCH 32
#define KOUT   256
#define POUT   110
#define QOUT   110
#define HW     (HIN * WIN)         // 12544
#define PQ     (POUT * QOUT)       // 12100
#define KG     (CIN * 9)           // 1152
#define BK     32
#define NSTEP  (KG / BK)           // 36
#define BM     128                 // output-channel tile
#define BN     128                 // pixel tile
#define LDSTR  40                  // LDS row stride in halves (80B -> conflict-free b128)

typedef __attribute__((ext_vector_type(16))) __bf16 v16bf;
typedef __attribute__((ext_vector_type(8)))  __bf16 v8bf;
typedef __attribute__((ext_vector_type(8)))  float  v8f;

union FragAB { v16bf v; v8bf h[2]; };

__device__ __forceinline__ unsigned int f2bf1(float f) {
  unsigned int u = __float_as_uint(f);
  u += 0x7FFFu + ((u >> 16) & 1u);   // round-to-nearest-even (prep only)
  return u >> 16;
}

// Hot-path pack: round-half-up + single v_perm_b32 byte-pack.
__device__ __forceinline__ unsigned int pack2bf(float lo, float hi) {
  const unsigned a = __float_as_uint(lo) + 0x8000u;
  const unsigned b = __float_as_uint(hi) + 0x8000u;
  // result = {b.byte3, b.byte2, a.byte3, a.byte2}; sel 0-3 -> arg1, 4-7 -> arg0
  return __builtin_amdgcn_perm(b, a, 0x07060302u);
}

// ---------------------------------------------------------------------------
// Prep: filt OIHW f32 -> wsA bf16 [o][k'] with k' = rs*128 + c.
// ---------------------------------------------------------------------------
__global__ __launch_bounds__(256) void filt_to_bf16(
    const float* __restrict__ filt, unsigned int* __restrict__ wsA) {
  const int t  = blockIdx.x * 256 + threadIdx.x;  // 0 .. 147455, exact
  const int o  = t / 576;
  const int p2 = t - o * 576;       // pair index within row = rs*64 + c/2
  const int rs = p2 >> 6;
  const int c  = (p2 & 63) * 2;
  const float f0 = filt[(o * CIN + c) * 9 + rs];
  const float f1 = filt[(o * CIN + c + 1) * 9 + rs];
  wsA[o * 576 + p2] = f2bf1(f0) | (f2bf1(f1) << 16);
}

// ---------------------------------------------------------------------------
// Main implicit-GEMM kernel.
// ---------------------------------------------------------------------------
__global__ __launch_bounds__(256) void conv_igemm_bf16(
    const float* __restrict__ x, const unsigned int* __restrict__ wsA,
    float* __restrict__ out) {
  __shared__ __align__(16) unsigned short As[2][BM * LDSTR]; // [o][k] bf16 bits
  __shared__ __align__(16) unsigned short Bs[2][BN * LDSTR]; // [pix][k] bf16 bits

  const int tid   = threadIdx.x;
  const int mTile = blockIdx.x;   // pixel tile, 0..3024
  const int oTile = blockIdx.y;   // channel tile, 0..1

  // ---------------- loader thread mapping ----------------
  const int aRow  = tid & (BM - 1);     // channel row within tile
  const int bPix  = tid & (BN - 1);     // pixel within tile
  const int half  = tid >> 7;           // 0 or 1: which 16-wide k half
  const int kHalf = half * 16;

  const int  mLoad = mTile * BN + bPix;
  const int  nImgL = mLoad / PQ;
  const int  remL  = mLoad - nImgL * PQ;
  const int  pL    = remL / QOUT;
  const int  qL    = remL - pL * QOUT;
  const float* gB  = x + (long)nImgL * CIN * HW + (long)pL * WIN + qL;

  // This thread's 32-byte slice of the bf16 filter row (uints): row*576 + step*16 + half*8
  const unsigned long long gABase =
      (unsigned long long)(const void*)(wsA + (long)(oTile * BM + aRow) * 576 + half * 8);

  float bReg[16];        // B staging: 16 f32 -> pack on stage

  // A tile: async DMA straight into LDS (32 B per thread = 2 x b128).
  auto asyncA = [&](int buf, int step) {
    const unsigned ldsA =
        (unsigned)(uintptr_t)&As[buf][aRow * LDSTR + kHalf];
    const unsigned long long ga = gABase + (unsigned long long)step * 64ull;
    asm volatile(
        "global_load_async_to_lds_b128 %0, %1, off\n\t"
        "global_load_async_to_lds_b128 %0, %1, off offset:16"
        :: "v"(ldsA), "v"(ga) : "memory");
  };

  auto fetchB = [&](int step) {
    const int rs = step >> 2;          // fixed (r,s) for whole step
    const int c0 = (step & 3) * BK;
    const int r  = rs / 3;
    const int s  = rs - r * 3;
    const float* p = gB + (long)(c0 + kHalf) * HW + r * WIN + s;
#pragma unroll
    for (int j = 0; j < 16; ++j)       // immediate offsets j*HW*4 < 24-bit
      bReg[j] = p[j * HW];
  };

  auto stageB = [&](int buf) {
    unsigned int bp[8];
#pragma unroll
    for (int i = 0; i < 8; ++i)
      bp[i] = pack2bf(bReg[2 * i], bReg[2 * i + 1]);
    uint4* dstB = (uint4*)&Bs[buf][bPix * LDSTR + kHalf];
    dstB[0] = make_uint4(bp[0], bp[1], bp[2], bp[3]);
    dstB[1] = make_uint4(bp[4], bp[5], bp[6], bp[7]);
  };

  // ---------------- compute thread mapping ----------------
  const int lane = tid & 31;
  const int wave = tid >> 5;
  const int wm   = wave & 1;    // channel-dir wave (x2 -> 64 rows each)
  const int wn   = wave >> 1;   // pixel-dir wave   (x4 -> 32 cols each)
  const int lrow = lane & 15;
  const int hsel = lane >> 4;   // half-wave select

  v8f acc[4][2];
  const v8f vzero = {0.f, 0.f, 0.f, 0.f, 0.f, 0.f, 0.f, 0.f};
#pragma unroll
  for (int i = 0; i < 4; ++i)
#pragma unroll
    for (int j = 0; j < 2; ++j) acc[i][j] = vzero;

  auto compute = [&](int buf) {
    FragAB a[4], b[2];
#pragma unroll
    for (int fm = 0; fm < 4; ++fm) {
      const int row = (wm * 64 + fm * 16 + lrow) * LDSTR;
      // A: lanes 0-15 hold K 0..15, lanes 16-31 hold K 16..31 (contiguous 32B, matches B)
      a[fm].h[0] = *(const v8bf*)&As[buf][row + hsel * 16];
      a[fm].h[1] = *(const v8bf*)&As[buf][row + hsel * 16 + 8];
    }
#pragma unroll
    for (int fn = 0; fn < 2; ++fn) {
      const int col = (wn * 32 + fn * 16 + lrow) * LDSTR;
      // B: lanes 0-15 hold K 0..15, lanes 16-31 hold K 16..31 (contiguous 32B)
      b[fn].h[0] = *(const v8bf*)&Bs[buf][col + hsel * 16];
      b[fn].h[1] = *(const v8bf*)&Bs[buf][col + hsel * 16 + 8];
    }
#pragma unroll
    for (int fm = 0; fm < 4; ++fm)
#pragma unroll
      for (int fn = 0; fn < 2; ++fn)
        acc[fm][fn] = __builtin_amdgcn_wmma_f32_16x16x32_bf16(
            false, a[fm].v, false, b[fn].v, (short)0, acc[fm][fn], false, false);
  };

  // ---------------- pipelined main loop ----------------
  asyncA(0, 0);
  fetchB(0);
  stageB(0);
  asm volatile("s_wait_asynccnt 0x0" ::: "memory");
  __syncthreads();
#pragma unroll 1
  for (int step = 0; step < NSTEP; ++step) {
    const int cur = step & 1;
    if (step + 1 < NSTEP) {
      asyncA(cur ^ 1, step + 1);       // DMA next A tile into LDS
      fetchB(step + 1);                // VMEM overlaps WMMA below
    }
    compute(cur);
    if (step + 1 < NSTEP) stageB(cur ^ 1);
    asm volatile("s_wait_asynccnt 0x0" ::: "memory");
    __syncthreads();
  }

  // ---------------- epilogue: coalesced along q ----------------
#pragma unroll
  for (int fn = 0; fn < 2; ++fn) {
    const int m    = mTile * BN + wn * 32 + fn * 16 + lrow;
    const int nImg = m / PQ;
    const int rem  = m - nImg * PQ;
    const int p    = rem / QOUT;
    const int q    = rem - p * QOUT;
#pragma unroll
    for (int fm = 0; fm < 4; ++fm) {
      const int oB = oTile * BM + wm * 64 + fm * 16 + hsel * 8;
      float* dst = out + (((long)nImg * KOUT + oB) * POUT + p) * QOUT + q;
#pragma unroll
      for (int r = 0; r < 8; ++r)
        dst[(long)r * PQ] = acc[fm][fn][r];
    }
  }
}

extern "C" void kernel_launch(void* const* d_in, const int* in_sizes, int n_in,
                              void* d_out, int out_size, void* d_ws, size_t ws_size,
                              hipStream_t stream) {
  (void)in_sizes; (void)n_in; (void)ws_size; (void)out_size;
  const float* x    = (const float*)d_in[0];
  const float* filt = (const float*)d_in[1];
  float*       out  = (float*)d_out;
  unsigned int* wsA = (unsigned int*)d_ws;   // 256*1152 bf16 = 589,824 bytes

  filt_to_bf16<<<576, 256, 0, stream>>>(filt, wsA);
  dim3 grid(3025, 2, 1);
  conv_igemm_bf16<<<grid, 256, 0, stream>>>(x, wsA, out);
}
